// ChannelAttention_6691559047247
// MI455X (gfx1250) — matrix operand order repaired and checked by reading, not compiled, not run on hardware
//
#include <hip/hip_runtime.h>
#include <hip/hip_bf16.h>

// ---------------------------------------------------------------------------
// Channel attention for MI455X (gfx1250), wave32 + v_wmma_f32_16x16x32_bf16.
//
// attn[d,e] = softmax_e( exp(scale) * (q^T k)[d,e] / (||q_d|| ||k_e||) )
// so we compute the raw 32x32 Gram per (b,head) and fold the token-axis
// L2 norms in afterwards -- no normalized q/k tensors are materialized.
// ---------------------------------------------------------------------------

typedef __attribute__((ext_vector_type(16))) __bf16 v16bf;
typedef __attribute__((ext_vector_type(8)))  __bf16 v8bf;
typedef __attribute__((ext_vector_type(8)))  float  v8f;

#define B_   8
#define NH   4
#define HD   32
#define C_   128
#define N_   16384          // tokens per batch image (128*128)
#define BN_  (B_ * N_)      // 131072
#define EPS_ 1.55e-05f

// ---- WMMA fragment helpers (contiguous-K order, same for A and B, wave32) ----------
// A (16x32 bf16): lane L holds row M=L%16, K run of 16 at 16*(L/16).
__device__ __forceinline__ v16bf load_a_frag(const __bf16* p, int ld) {
  const int lane = threadIdx.x & 31;
  const int row  = lane & 15;
  const int ko   = (lane >> 4) * 16;
  const __bf16* q = p + (size_t)row * ld + ko;
  v16bf a;
  ((v8bf*)&a)[0] = *(const v8bf*)(q);
  ((v8bf*)&a)[1] = *(const v8bf*)(q + 8);
  return a;
}
// B (32x16 bf16): lane L holds col N=L%16, K run of 16 at 16*(L/16).
// Source must be column-major for B, i.e. stored as [col][k], row stride ld.
__device__ __forceinline__ v16bf load_b_frag(const __bf16* p, int ld) {
  const int lane = threadIdx.x & 31;
  const int col  = lane & 15;
  const int ko   = (lane >> 4) * 16;
  return *(const v16bf*)(p + (size_t)col * ld + ko);
}

__device__ __forceinline__ v8f wmma_bf16(v16bf a, v16bf b, v8f c) {
  return __builtin_amdgcn_wmma_f32_16x16x32_bf16(
      /*neg_a=*/false, a, /*neg_b=*/false, b,
      /*c_mod=*/(short)0, c, /*reuse_a=*/false, /*reuse_b=*/false);
}

// ---- tiny conversion kernels ----------------------------------------------
__global__ __launch_bounds__(256) void cvt_bf16(const float* __restrict__ in,
                                                __bf16* __restrict__ out, int n) {
  int i = blockIdx.x * 256 + threadIdx.x;
  if (i < n) out[i] = (__bf16)in[i];
}
// in: [rows][cols] f32 row-major -> out: [cols][rows] bf16 (transposed)
__global__ __launch_bounds__(256) void transpose_bf16(const float* __restrict__ in,
                                                      __bf16* __restrict__ out,
                                                      int rows, int cols) {
  int i = blockIdx.x * 256 + threadIdx.x;
  if (i < rows * cols) {
    int r = i / cols, c = i % cols;
    out[(size_t)c * rows + r] = (__bf16)in[i];
  }
}

// ---- kernel 1: qkv = x @ W + bias; q,k stored channel-major, v token-major -
// One 32x32 output tile per wave (2x2 WMMA), K = 128 in 4 steps.
__global__ __launch_bounds__(256) void qkv_gemm(
    const __bf16* __restrict__ xb,    // [BN][128]
    const __bf16* __restrict__ wT,    // [384][128]  (qkv_w transposed)
    const float*  __restrict__ q_bias,
    const float*  __restrict__ v_bias,
    __bf16* __restrict__ qT,          // [B*128][N]  (b, h*32+d, n)
    __bf16* __restrict__ kT,          // [B*128][N]
    __bf16* __restrict__ vN)          // [BN][128]
{
  const int lane   = threadIdx.x & 31;
  const int waveId = blockIdx.x * 8 + (threadIdx.x >> 5);
  const int tileCol = waveId % 12;          // 384/32
  const int tileRow = waveId / 12;          // BN/32 = 4096
  const int row0 = tileRow * 32;
  const int col0 = tileCol * 32;

  v8f acc[2][2] = {};
  for (int k0 = 0; k0 < C_; k0 += 32) {
    v16bf a0 = load_a_frag(xb + (size_t)row0 * C_ + k0, C_);
    v16bf a1 = load_a_frag(xb + (size_t)(row0 + 16) * C_ + k0, C_);
    v16bf b0 = load_b_frag(wT + (size_t)col0 * C_ + k0, C_);
    v16bf b1 = load_b_frag(wT + (size_t)(col0 + 16) * C_ + k0, C_);
    acc[0][0] = wmma_bf16(a0, b0, acc[0][0]);
    acc[0][1] = wmma_bf16(a0, b1, acc[0][1]);
    acc[1][0] = wmma_bf16(a1, b0, acc[1][0]);
    acc[1][1] = wmma_bf16(a1, b1, acc[1][1]);
  }

  const int bIdx  = row0 / N_;
  const int nloc0 = row0 % N_;
  for (int ni = 0; ni < 2; ++ni) {
    const int col = col0 + ni * 16 + (lane & 15);
    float bias = 0.f;
    if (col < 128)       bias = q_bias[col];
    else if (col >= 256) bias = v_bias[col - 256];
    for (int mi = 0; mi < 2; ++mi) {
      v8f v = acc[mi][ni];
      const int tok0 = nloc0 + mi * 16 + (lane >> 4) * 8;  // 8 consecutive rows
      if (col < 256) {
        // transposed (channel-major) packed 16B store
        __bf16* dst = (col < 128) ? qT : kT;
        const int cc = (col < 128) ? col : col - 128;
        v8bf pk;
#pragma unroll
        for (int r = 0; r < 8; ++r) pk[r] = (__bf16)(v[r] + bias);
        *(v8bf*)(dst + ((size_t)(bIdx * 128 + cc)) * N_ + tok0) = pk;
      } else {
        const int vc = col - 256;
#pragma unroll
        for (int r = 0; r < 8; ++r)
          vN[((size_t)(bIdx * N_ + tok0 + r)) * C_ + vc] = (__bf16)(v[r] + bias);
      }
    }
  }
}

// ---- kernel 2: per-column (token-axis) sum of squares for q and k ---------
// qT and kT are adjacent in ws: 8192 columns of 16384 bf16. One wave/column.
__global__ __launch_bounds__(256) void col_sumsq(const __bf16* __restrict__ qkT,
                                                 float* __restrict__ sq) {
  const int col  = blockIdx.x * 8 + (threadIdx.x >> 5);
  const int lane = threadIdx.x & 31;
  const __bf16* p = qkT + (size_t)col * N_;
  float s = 0.f;
  for (int i = lane * 8; i < N_; i += 32 * 8) {
    v8bf v = *(const v8bf*)(p + i);
#pragma unroll
    for (int r = 0; r < 8; ++r) { float f = (float)v[r]; s = fmaf(f, f, s); }
  }
#pragma unroll
  for (int off = 16; off; off >>= 1) s += __shfl_xor(s, off, 32);
  if (lane == 0) sq[col] = s;
}

// ---- kernel 3: per (b,h): Gram -> scaled/normalized softmax -> attn @ v ----
__global__ __launch_bounds__(256) void chan_attn(
    const __bf16* __restrict__ qT,
    const __bf16* __restrict__ kT,
    const __bf16* __restrict__ vN,     // [BN][128]
    const float*  __restrict__ sq,     // [8192] (q cols, then k cols)
    const float*  __restrict__ scale,  // [4]
    __bf16* __restrict__ outT)         // [BN][128]
{
  __shared__ __align__(16) float  G[32][32];
  __shared__ __align__(16) float  rqs[32], rks[32];
  __shared__ __align__(16) __bf16 attnS[32][32];
  __shared__ float esS;

  const int bh   = blockIdx.x;            // b*4 + h
  const int bIdx = bh >> 2, h = bh & 3;
  const int tid  = threadIdx.x, lane = tid & 31, wave = tid >> 5;

  for (int i = tid; i < 32 * 32; i += 256) ((float*)G)[i] = 0.f;
  __syncthreads();

  const __bf16* qbase = qT + (size_t)(bIdx * 128 + h * HD) * N_;
  const __bf16* kbase = kT + (size_t)(bIdx * 128 + h * HD) * N_;

  // Phase 1: G = q^T k over this wave's 2048-token chunk (K-split, 8 waves)
  v8f acc[2][2] = {};
  const int n0 = wave * 2048;
  for (int t = n0; t < n0 + 2048; t += 32) {
    v16bf a0 = load_a_frag(qbase + t, N_);            // rows d = 0..15
    v16bf a1 = load_a_frag(qbase + 16 * N_ + t, N_);  // rows d = 16..31
    v16bf b0 = load_b_frag(kbase + t, N_);            // cols e = 0..15
    v16bf b1 = load_b_frag(kbase + 16 * N_ + t, N_);  // cols e = 16..31
    acc[0][0] = wmma_bf16(a0, b0, acc[0][0]);
    acc[0][1] = wmma_bf16(a0, b1, acc[0][1]);
    acc[1][0] = wmma_bf16(a1, b0, acc[1][0]);
    acc[1][1] = wmma_bf16(a1, b1, acc[1][1]);
  }
#pragma unroll
  for (int mi = 0; mi < 2; ++mi)
#pragma unroll
    for (int ni = 0; ni < 2; ++ni) {
      const int d0 = mi * 16 + (lane >> 4) * 8;   // D: lanes 16-31 hold M=r+8
      const int e  = ni * 16 + (lane & 15);
#pragma unroll
      for (int r = 0; r < 8; ++r) atomicAdd(&G[d0 + r][e], acc[mi][ni][r]);
    }
  __syncthreads();

  // Phase 2: fold in norms + exp(scale), 32-wide softmax, stash attn as bf16
  if (tid < 32) {
    rqs[tid] = rsqrtf(fmaxf(sq[bh * 32 + tid], EPS_));
    rks[tid] = rsqrtf(fmaxf(sq[4096 + bh * 32 + tid], EPS_));
    if (tid == 0) esS = __expf(scale[h]);
  }
  __syncthreads();
  if (tid < 32) {
    const int d = tid;
    const float es = esS * rqs[d];
    float row[32];
    float mx = -1e30f;
#pragma unroll
    for (int e = 0; e < 32; ++e) { row[e] = es * rks[e] * G[d][e]; mx = fmaxf(mx, row[e]); }
    float sum = 0.f;
#pragma unroll
    for (int e = 0; e < 32; ++e) { row[e] = __expf(row[e] - mx); sum += row[e]; }
    const float inv = 1.f / sum;
#pragma unroll
    for (int e = 0; e < 32; ++e) attnS[d][e] = (__bf16)(row[e] * inv);
  }
  __syncthreads();

  // Phase 3: out[d, n] = sum_e attn[d,e] * v[n,e]; write (n, h*32+d) bf16
  v16bf aA0 = load_a_frag(&attnS[0][0], 32);
  v16bf aA1 = load_a_frag(&attnS[16][0], 32);
  const __bf16* vbase = vN   + (size_t)(bIdx * N_) * C_ + h * HD;
  __bf16*       obase = outT + (size_t)(bIdx * N_) * C_ + h * HD;
  for (int t = n0; t < n0 + 2048; t += 16) {
    v16bf bv = load_b_frag(vbase + (size_t)t * C_, C_);   // B[k=e, col=n]
    v8f o0 = {}, o1 = {};
    o0 = wmma_bf16(aA0, bv, o0);
    o1 = wmma_bf16(aA1, bv, o1);
    const size_t rbase = (size_t)(t + (lane & 15)) * C_;  // fixed token col
    v8bf p0, p1;
#pragma unroll
    for (int r = 0; r < 8; ++r) { p0[r] = (__bf16)o0[r]; p1[r] = (__bf16)o1[r]; }
    *(v8bf*)(obase + rbase + 0  + (lane >> 4) * 8) = p0;  // d = 0..15 block
    *(v8bf*)(obase + rbase + 16 + (lane >> 4) * 8) = p1;  // d = 16..31 block
  }
}

// ---- kernel 4: final projection  out = out_t @ proj_w + proj_b (f32 out) ---
__global__ __launch_bounds__(256) void proj_gemm(
    const __bf16* __restrict__ outT,   // [BN][128] bf16
    const __bf16* __restrict__ wpT,    // [128][128] bf16 (proj_w transposed)
    const float*  __restrict__ proj_b,
    float* __restrict__ out)           // [BN][128] f32
{
  const int lane   = threadIdx.x & 31;
  const int waveId = blockIdx.x * 8 + (threadIdx.x >> 5);
  const int tileCol = waveId & 3;            // 128/32
  const int tileRow = waveId >> 2;           // 4096
  const int row0 = tileRow * 32, col0 = tileCol * 32;

  v8f acc[2][2] = {};
  for (int k0 = 0; k0 < C_; k0 += 32) {
    v16bf a0 = load_a_frag(outT + (size_t)row0 * C_ + k0, C_);
    v16bf a1 = load_a_frag(outT + (size_t)(row0 + 16) * C_ + k0, C_);
    v16bf b0 = load_b_frag(wpT + (size_t)col0 * C_ + k0, C_);
    v16bf b1 = load_b_frag(wpT + (size_t)(col0 + 16) * C_ + k0, C_);
    acc[0][0] = wmma_bf16(a0, b0, acc[0][0]);
    acc[0][1] = wmma_bf16(a0, b1, acc[0][1]);
    acc[1][0] = wmma_bf16(a1, b0, acc[1][0]);
    acc[1][1] = wmma_bf16(a1, b1, acc[1][1]);
  }
  for (int ni = 0; ni < 2; ++ni) {
    const int col = col0 + ni * 16 + (lane & 15);
    const float bias = proj_b[col];
    for (int mi = 0; mi < 2; ++mi) {
      const int r0 = row0 + mi * 16 + (lane >> 4) * 8;
#pragma unroll
      for (int r = 0; r < 8; ++r)
        out[(size_t)(r0 + r) * C_ + col] = acc[mi][ni][r] + bias;
    }
  }
}

// ---------------------------------------------------------------------------
extern "C" void kernel_launch(void* const* d_in, const int* in_sizes, int n_in,
                              void* d_out, int out_size, void* d_ws, size_t ws_size,
                              hipStream_t stream) {
  const float* x      = (const float*)d_in[0];   // [8,128,128,128]
  const float* qkv_w  = (const float*)d_in[1];   // [128,384]
  const float* q_bias = (const float*)d_in[2];   // [128]
  const float* v_bias = (const float*)d_in[3];   // [128]
  const float* scale  = (const float*)d_in[4];   // [4,1,1]
  const float* proj_w = (const float*)d_in[5];   // [128,128]
  const float* proj_b = (const float*)d_in[6];   // [128]
  float* out = (float*)d_out;

  char* ws = (char*)d_ws;
  const size_t SZ_QT = (size_t)B_ * 128 * N_ * sizeof(__bf16);   // 32 MiB
  __bf16* qT     = (__bf16*)(ws);                     // [B*128][N]
  __bf16* kT     = (__bf16*)(ws + SZ_QT);             // [B*128][N] (adjacent!)
  __bf16* vN     = (__bf16*)(ws + 2 * SZ_QT);         // [BN][128]
  __bf16* xb     = (__bf16*)(ws + 3 * SZ_QT);         // [BN][128]
  __bf16* outT   = xb;                                // alias: xb dead after qkv_gemm
  __bf16* wqkvT  = (__bf16*)(ws + 4 * SZ_QT);         // [384][128]
  __bf16* wprojT = (__bf16*)(ws + 4 * SZ_QT + 384 * 128 * sizeof(__bf16));
  float*  sq     = (float*)(ws + 4 * SZ_QT + (384 + 128) * 128 * sizeof(__bf16)); // [8192]

  // 0) precision conversion + weight transposes (B-fragments want [out][in])
  cvt_bf16<<<(BN_ * C_) / 256, 256, 0, stream>>>(x, xb, BN_ * C_);
  transpose_bf16<<<(128 * 384) / 256, 256, 0, stream>>>(qkv_w, wqkvT, 128, 384);
  transpose_bf16<<<(128 * 128) / 256, 256, 0, stream>>>(proj_w, wprojT, 128, 128);

  // 1) QKV GEMM: 49152 wave-tiles of 32x32, 8 waves / WG
  qkv_gemm<<<6144, 256, 0, stream>>>(xb, wqkvT, q_bias, v_bias, qT, kT, vN);

  // 2) token-axis sum-of-squares for q and k columns (8192 columns)
  col_sumsq<<<1024, 256, 0, stream>>>(qT, sq);

  // 3) per-(b,head) Gram + softmax + attn@v
  chan_attn<<<B_ * NH, 256, 0, stream>>>(qT, kT, vN, sq, scale, outT);

  // 4) output projection
  proj_gemm<<<2048, 256, 0, stream>>>(outT, wprojT, proj_b, out);
}
